// BiResBlock_1778116460658
// MI455X (gfx1250) — 8-bit matrix operand order repaired and checked by reading, compiled, not run on hardware
//
#include <hip/hip_runtime.h>
#include <stdint.h>


typedef __attribute__((ext_vector_type(8))) int   v8i;
typedef __attribute__((ext_vector_type(4))) int   v4i;
typedef __attribute__((ext_vector_type(2))) int   v2i;

#define NB    32
#define CH    256
#define HH    56
#define WWD   56
#define HP    58
#define WP    58
#define HWSZ  (HH * WWD)
#define NHW   (NB * HWSZ)
#define NELEM ((size_t)NB * CH * HWSZ)
#define ACTSZ ((size_t)NB * HP * WP * CH)
#define BN_EPS 1e-5f
#define LDSROW 80

static constexpr size_t OFF_WPACK = 0;
static constexpr size_t OFF_SCALE = 589824;
static constexpr size_t OFF_SUM   = OFF_SCALE + 1024;
static constexpr size_t OFF_SUMSQ = OFF_SUM   + 1024;
static constexpr size_t OFF_ACT   = OFF_SUMSQ + 1024;
static constexpr size_t OFF_Y     = ((OFF_ACT + ACTSZ + 511) / 512) * 512;

__device__ __forceinline__ void block_sync_lds() {
    asm volatile(
        "s_wait_dscnt 0x0\n\t"
        "s_barrier_signal -1\n\t"
        "s_barrier_wait -1"
        ::: "memory");
}

struct AFrag { v2i p0, p1, p2, p3; };

__device__ __forceinline__ AFrag lda(const int8_t* p) {
    AFrag f;
    f.p0 = *(const v2i*)(p +  0);
    f.p1 = *(const v2i*)(p +  8);
    f.p2 = *(const v2i*)(p + 32);
    f.p3 = *(const v2i*)(p + 40);
    return f;
}
__device__ __forceinline__ v8i mka(const AFrag& f) {
    v8i A;
    A[0] = f.p0[0]; A[1] = f.p0[1]; A[2] = f.p1[0]; A[3] = f.p1[1];
    A[4] = f.p2[0]; A[5] = f.p2[1]; A[6] = f.p3[0]; A[7] = f.p3[1];
    return A;
}
__device__ __forceinline__ v8i mkb(const v4i lo, const v4i hi) {
    v8i B;
    B[0] = lo[0]; B[1] = lo[1]; B[2] = lo[2]; B[3] = lo[3];
    B[4] = hi[0]; B[5] = hi[1]; B[6] = hi[2]; B[7] = hi[3];
    return B;
}

__global__ void zero_stats_kernel(float* sums) {
    sums[threadIdx.x] = 0.0f;
}

__global__ void zero_act_kernel(v4i* act16) {
    const v4i z = {0, 0, 0, 0};
    act16[(size_t)blockIdx.x * 256 + threadIdx.x] = z;
}

__global__ void prep_weights_kernel(const float* __restrict__ w,
                                    int8_t* __restrict__ wpack,
                                    float* __restrict__ scale) {
    const int co = blockIdx.x;
    const int ci = threadIdx.x;
    const float* wp = w + ((size_t)co * CH + ci) * 9;
    float asum = 0.0f;
#pragma unroll
    for (int t = 0; t < 9; ++t) {
        float v = wp[t];
        asum += fabsf(v);
        wpack[(size_t)t * CH * CH + (size_t)co * CH + ci] =
            (v > 0.0f) ? (int8_t)1 : ((v < 0.0f) ? (int8_t)-1 : (int8_t)0);
    }
    __shared__ float red[256];
    red[ci] = asum;
    __syncthreads();
    for (int s = 128; s > 0; s >>= 1) {
        if (ci < s) red[ci] += red[ci + s];
        __syncthreads();
    }
    if (ci == 0) scale[co] = red[0] / 2304.0f;
}

__global__ void binact_kernel(const float* __restrict__ x,
                              const float* __restrict__ bias,
                              int8_t* __restrict__ act) {
    const int idx = blockIdx.x * 256 + threadIdx.x;
    const int c  = idx & 255;
    const int sp = idx >> 8;
    const int wq = sp % WWD;
    const int t  = sp / WWD;
    const int hq = t % HH;
    const int nq = t / HH;
    const float s = x[((size_t)(nq * CH + c)) * HWSZ + hq * WWD + wq] + bias[c];
    act[((size_t)((nq * HP + hq + 1) * WP + wq + 1)) * CH + c] =
        (s > 0.0f) ? (int8_t)1 : ((s < 0.0f) ? (int8_t)-1 : (int8_t)0);
}

__global__ void __launch_bounds__(256, 2)
bconv_gemm_kernel(const int8_t* __restrict__ wpack,
                  const int8_t* __restrict__ act,
                  const float*  __restrict__ scale,
                  float* __restrict__ y,
                  float* __restrict__ sum,
                  float* __restrict__ sumsq) {
    __shared__ __align__(16) int8_t ldsB[2][32 * LDSROW];

    const int tid  = threadIdx.x;
    const int wave = tid >> 5;
    const int lane = tid & 31;
    const int half = lane >> 4;
    const int l16  = lane & 15;

    const int npair = blockIdx.x;
    const int mpair = wave;

    const int col0 = npair * 32 + l16;
    const int col1 = col0 + 16;
    const int wq0 = col0 % WWD, t0 = col0 / WWD, hq0 = t0 % HH, nq0 = t0 / HH;
    const int wq1 = col1 % WWD, t1 = col1 / WWD, hq1 = t1 % HH, nq1 = t1 / HH;

    const int ccol = tid >> 3;
    const int cseg = tid & 7;
    const int colg = npair * 32 + ccol;
    const int wqc = colg % WWD, tc = colg / WWD, hqc = tc % HH, nqc = tc / HH;
    const int8_t* cbase =
        act + ((size_t)((nqc * HP + hqc) * WP + wqc)) * CH + cseg * 8;
    const int ldsoff = ccol * LDSROW + cseg * 8;

    const int m0   = mpair * 32 + l16;
    const int aklo = half * 16;
    const int bklo = half * 16;
    const int8_t* abase0 = wpack + (size_t)m0 * CH + aklo;
    const int8_t* abase1 = abase0 + (size_t)16 * CH;

    v8i acc00 = {0,0,0,0,0,0,0,0};
    v8i acc01 = {0,0,0,0,0,0,0,0};
    v8i acc10 = {0,0,0,0,0,0,0,0};
    v8i acc11 = {0,0,0,0,0,0,0,0};

    v2i   gB  = *(const v2i*)(cbase);
    AFrag A0c = lda(abase0);
    AFrag A1c = lda(abase1);

#pragma unroll
    for (int step = 0; step < 36; ++step) {
        v2i   gBn{};
        AFrag A0n{}, A1n{};
        if (step < 35) {
            const int ntap = (step + 1) >> 2;
            const int nkc  = ((step + 1) & 3) << 6;
            const int ndh  = ntap / 3, ndw = ntap % 3;
            gBn = *(const v2i*)(cbase + (size_t)(ndh * WP + ndw) * CH + nkc);
            A0n = lda(abase0 + (size_t)ntap * CH * CH + nkc);
            A1n = lda(abase1 + (size_t)ntap * CH * CH + nkc);
        }

        const int b = step & 1;
        *(v2i*)&ldsB[b][ldsoff] = gB;
        block_sync_lds();

        const v4i b0lo = *(const v4i*)&ldsB[b][l16 * LDSROW + bklo];
        const v4i b0hi = *(const v4i*)&ldsB[b][l16 * LDSROW + bklo + 32];
        const v4i b1lo = *(const v4i*)&ldsB[b][(16 + l16) * LDSROW + bklo];
        const v4i b1hi = *(const v4i*)&ldsB[b][(16 + l16) * LDSROW + bklo + 32];

        const v8i A0 = mka(A0c), A1 = mka(A1c);
        const v8i B0 = mkb(b0lo, b0hi), B1 = mkb(b1lo, b1hi);

        acc00 = __builtin_amdgcn_wmma_i32_16x16x64_iu8(true, A0, true, B0, acc00, false, false);
        acc10 = __builtin_amdgcn_wmma_i32_16x16x64_iu8(true, A1, true, B0, acc10, false, false);
        acc01 = __builtin_amdgcn_wmma_i32_16x16x64_iu8(true, A0, true, B1, acc01, false, false);
        acc11 = __builtin_amdgcn_wmma_i32_16x16x64_iu8(true, A1, true, B1, acc11, false, false);

        gB = gBn; A0c = A0n; A1c = A1n;
    }

    const int mb0 = mpair * 32 + half * 8;
    const int mb1 = mb0 + 16;
#pragma unroll
    for (int v = 0; v < 8; ++v) {
        const int r0 = mb0 + v;
        const int r1 = mb1 + v;
        const float s0 = scale[r0];
        const float s1 = scale[r1];

        const float yf00 = (float)acc00[v] * s0;
        const float yf01 = (float)acc01[v] * s0;
        const float yf10 = (float)acc10[v] * s1;
        const float yf11 = (float)acc11[v] * s1;

        y[((size_t)(nq0 * CH + r0)) * HWSZ + hq0 * WWD + wq0] = yf00;
        y[((size_t)(nq1 * CH + r0)) * HWSZ + hq1 * WWD + wq1] = yf01;
        y[((size_t)(nq0 * CH + r1)) * HWSZ + hq0 * WWD + wq0] = yf10;
        y[((size_t)(nq1 * CH + r1)) * HWSZ + hq1 * WWD + wq1] = yf11;

        float sA = yf00 + yf01, qA = yf00 * yf00 + yf01 * yf01;
        float sB = yf10 + yf11, qB = yf10 * yf10 + yf11 * yf11;
#pragma unroll
        for (int off = 8; off >= 1; off >>= 1) {
            sA += __shfl_xor(sA, off, 16);
            qA += __shfl_xor(qA, off, 16);
            sB += __shfl_xor(sB, off, 16);
            qB += __shfl_xor(qB, off, 16);
        }
        if (l16 == 0) {
            atomicAdd(&sum[r0],   sA);
            atomicAdd(&sumsq[r0], qA);
            atomicAdd(&sum[r1],   sB);
            atomicAdd(&sumsq[r1], qB);
        }
    }
}

__global__ void finalize_kernel(const float* __restrict__ y,
                                const float* __restrict__ x,
                                const float* __restrict__ sum,
                                const float* __restrict__ sumsq,
                                const float* __restrict__ gamma,
                                const float* __restrict__ beta,
                                const float* __restrict__ mv1,
                                const float* __restrict__ prelu,
                                const float* __restrict__ mv2,
                                float* __restrict__ out) {
    const size_t idx = (size_t)blockIdx.x * 256 + threadIdx.x;
    const int c = (int)((idx / HWSZ) & 255);
    const float inv = 1.0f / (float)NHW;
    const float m   = sum[c] * inv;
    const float var = fmaxf(sumsq[c] * inv - m * m, 0.0f);
    const float g   = gamma[c] * rsqrtf(var + BN_EPS);
    float yv = (y[idx] - m) * g + beta[c] + x[idx] + mv1[c];
    yv = (yv >= 0.0f) ? yv : prelu[c] * yv;
    out[idx] = yv + mv2[c];
}

extern "C" void kernel_launch(void* const* d_in, const int* in_sizes, int n_in,
                              void* d_out, int out_size, void* d_ws, size_t ws_size,
                              hipStream_t stream) {
    const float* x      = (const float*)d_in[0];
    const float* b0     = (const float*)d_in[1];
    const float* conv_w = (const float*)d_in[2];
    const float* gamma  = (const float*)d_in[3];
    const float* beta   = (const float*)d_in[4];
    const float* b1     = (const float*)d_in[5];
    const float* prelu  = (const float*)d_in[6];
    const float* b2     = (const float*)d_in[7];
    float* out = (float*)d_out;

    char* ws = (char*)d_ws;
    int8_t* wpack = (int8_t*)(ws + OFF_WPACK);
    float*  scale = (float*) (ws + OFF_SCALE);
    float*  sum   = (float*) (ws + OFF_SUM);
    float*  sumsq = (float*) (ws + OFF_SUMSQ);
    int8_t* act   = (int8_t*)(ws + OFF_ACT);
    float*  y     = (float*) (ws + OFF_Y);

    zero_stats_kernel<<<1, 512, 0, stream>>>(sum);
    zero_act_kernel<<<(unsigned)(ACTSZ / 16 / 256), 256, 0, stream>>>((v4i*)act);

    prep_weights_kernel<<<CH, CH, 0, stream>>>(conv_w, wpack, scale);

    binact_kernel<<<(unsigned)(NELEM / 256), 256, 0, stream>>>(x, b0, act);

    bconv_gemm_kernel<<<3136, 256, 0, stream>>>(wpack, act, scale,
                                                y, sum, sumsq);

    finalize_kernel<<<(unsigned)(NELEM / 256), 256, 0, stream>>>(
        y, x, sum, sumsq, gamma, beta, b1, prelu, b2, out);
}
